// CooperationModule_47330539602758
// MI455X (gfx1250) — compile-verified, matrix operand order checked by reading, not run on hardware
//
#include <hip/hip_runtime.h>
#include <hip/hip_bf16.h>
#include <stdint.h>

#define B_DIM 4096
#define E_DIM 16
#define D_DIM 512
#define H_DIM 2048

typedef __attribute__((ext_vector_type(16))) __bf16 v16bf;
typedef __attribute__((ext_vector_type(8)))  __bf16 v8bf;
typedef __attribute__((ext_vector_type(8)))  float  v8f;

// ---------- fp32 -> bf16 (round to nearest even) ----------
__device__ __forceinline__ unsigned short f32_to_bf16(float f) {
  union { float f; unsigned int u; } v; v.f = f;
  unsigned int u = v.u;
  u += 0x7FFFu + ((u >> 16) & 1u);
  return (unsigned short)(u >> 16);
}

__global__ void cvt_bf16_kernel(const float* __restrict__ src,
                                unsigned short* __restrict__ dst, int n4) {
  int i = blockIdx.x * blockDim.x + threadIdx.x;
  if (i >= n4) return;
  float4 f = reinterpret_cast<const float4*>(src)[i];
  ushort4 h;
  h.x = f32_to_bf16(f.x);
  h.y = f32_to_bf16(f.y);
  h.z = f32_to_bf16(f.z);
  h.w = f32_to_bf16(f.w);
  reinterpret_cast<ushort4*>(dst)[i] = h;
}

// ---------- beta[e,h] = bias[e,h] - dot(centers[e,:], W[e,h,:]) ----------
__global__ void beta_kernel(const float* __restrict__ centers,  // [E,D]
                            const float* __restrict__ W,        // [E,H,D]
                            const float* __restrict__ bias,     // [E,H]
                            float* __restrict__ beta) {         // [E,H]
  int idx = blockIdx.x * blockDim.x + threadIdx.x;  // e*H + h
  if (idx >= E_DIM * H_DIM) return;
  int e = idx >> 11;  // / H_DIM
  const float4* wrow = reinterpret_cast<const float4*>(W + (size_t)idx * D_DIM);
  const float4* crow = reinterpret_cast<const float4*>(centers + (size_t)e * D_DIM);
  float acc = 0.f;
#pragma unroll 8
  for (int d = 0; d < D_DIM / 4; ++d) {
    float4 w = wrow[d];
    float4 c = crow[d];
    acc += w.x * c.x + w.y * c.y + w.z * c.z + w.w * c.w;
  }
  beta[idx] = bias[idx] - acc;
}

// ---------- WMMA helpers ----------
__device__ __forceinline__ v16bf ldfrag(const unsigned short* p) {
  // p points at (row, k + kh); low 8 bf16 = K kh..kh+7, high 8 = K 16+kh..16+kh+7
  v8bf lo = *reinterpret_cast<const v8bf*>(p);
  v8bf hi = *reinterpret_cast<const v8bf*>(p + 16);
  return __builtin_shufflevector(lo, hi, 0, 1, 2, 3, 4, 5, 6, 7,
                                 8, 9, 10, 11, 12, 13, 14, 15);
}

__device__ __forceinline__ v8f wmma_bf16(v16bf a, v16bf b, v8f c) {
  return __builtin_amdgcn_wmma_f32_16x16x32_bf16(
      /*neg_a=*/false, a, /*neg_b=*/false, b,
      /*c_mod=*/(short)0, c, /*reuse_a=*/false, /*reuse_b=*/false);
}

// ---------- fused GEMM + relu-sum over experts ----------
// block = 256 threads (8 wave32), waves tiled 4(M) x 2(N).
// Each wave computes a 32x64 output sub-tile (2x4 WMMA tiles) -> block 128x128.
// 8 WMMA per k-step vs 12 b128 loads (1.5 loads/WMMA).
__global__ __launch_bounds__(256)
void moe_relu_sum_kernel(const unsigned short* __restrict__ xb,  // [B,D] bf16
                         const unsigned short* __restrict__ wb,  // [E,H,D] bf16
                         const float* __restrict__ beta,         // [E,H]
                         float* __restrict__ out) {              // [B,H]
  const int lane  = threadIdx.x & 31;
  const int wave  = threadIdx.x >> 5;
  const int waveM = wave & 3;   // 0..3 -> 32-row slices
  const int waveN = wave >> 2;  // 0..1 -> 64-col slices
  const int mBase = blockIdx.y * 128 + waveM * 32;
  const int nBase = blockIdx.x * 128 + waveN * 64;

  const int r  = lane & 15;        // row (A) / col (B) within 16-tile
  const int kh = (lane >> 4) * 8;  // K sub-offset per ISA 16-bit layout

  const unsigned short* aptr = xb + (size_t)(mBase + r) * D_DIM + kh;

  v8f s00 = {}, s01 = {}, s02 = {}, s03 = {};
  v8f s10 = {}, s11 = {}, s12 = {}, s13 = {};

  for (int e = 0; e < E_DIM; ++e) {
    const unsigned short* bptr =
        wb + ((size_t)e * H_DIM + nBase + r) * D_DIM + kh;

    v8f c00 = {}, c01 = {}, c02 = {}, c03 = {};
    v8f c10 = {}, c11 = {}, c12 = {}, c13 = {};

#pragma unroll 2
    for (int k = 0; k < D_DIM; k += 32) {
      v16bf a0 = ldfrag(aptr + k);
      v16bf a1 = ldfrag(aptr + 16 * D_DIM + k);
      v16bf b0 = ldfrag(bptr + k);
      v16bf b1 = ldfrag(bptr + 16 * D_DIM + k);
      v16bf b2 = ldfrag(bptr + 32 * D_DIM + k);
      v16bf b3 = ldfrag(bptr + 48 * D_DIM + k);
      c00 = wmma_bf16(a0, b0, c00);
      c01 = wmma_bf16(a0, b1, c01);
      c02 = wmma_bf16(a0, b2, c02);
      c03 = wmma_bf16(a0, b3, c03);
      c10 = wmma_bf16(a1, b0, c10);
      c11 = wmma_bf16(a1, b1, c11);
      c12 = wmma_bf16(a1, b2, c12);
      c13 = wmma_bf16(a1, b3, c13);
    }

    // beta is per output column (fixed per lane for all 8 C VGPRs)
    const float* bb = beta + e * H_DIM + nBase + r;
    float be0 = bb[0];
    float be1 = bb[16];
    float be2 = bb[32];
    float be3 = bb[48];
#pragma unroll
    for (int v = 0; v < 8; ++v) {
      s00[v] += fmaxf(c00[v] + be0, 0.0f);
      s01[v] += fmaxf(c01[v] + be1, 0.0f);
      s02[v] += fmaxf(c02[v] + be2, 0.0f);
      s03[v] += fmaxf(c03[v] + be3, 0.0f);
      s10[v] += fmaxf(c10[v] + be0, 0.0f);
      s11[v] += fmaxf(c11[v] + be1, 0.0f);
      s12[v] += fmaxf(c12[v] + be2, 0.0f);
      s13[v] += fmaxf(c13[v] + be3, 0.0f);
    }
  }

  // C/D layout: VGPR v, lane L -> row = v + 8*(L>=16), col = L&15
  const int colLo  = lane & 15;
  const int rowOff = (lane >> 4) * 8;
#pragma unroll
  for (int v = 0; v < 8; ++v) {
    float* o0 = out + (size_t)(mBase + rowOff + v) * H_DIM + nBase + colLo;
    float* o1 = o0 + (size_t)16 * H_DIM;
    o0[0]  = s00[v];
    o0[16] = s01[v];
    o0[32] = s02[v];
    o0[48] = s03[v];
    o1[0]  = s10[v];
    o1[16] = s11[v];
    o1[32] = s12[v];
    o1[48] = s13[v];
  }
}

extern "C" void kernel_launch(void* const* d_in, const int* in_sizes, int n_in,
                              void* d_out, int out_size, void* d_ws, size_t ws_size,
                              hipStream_t stream) {
  const float* x       = (const float*)d_in[0];  // [B,D]
  const float* centers = (const float*)d_in[1];  // [E,D]
  const float* W       = (const float*)d_in[2];  // [E,H,D]
  const float* bias    = (const float*)d_in[3];  // [E,H]
  float* out           = (float*)d_out;          // [B,H]

  // workspace layout
  char* ws = (char*)d_ws;
  unsigned short* xb   = (unsigned short*)ws;                                  // 4 MB
  unsigned short* wb   = (unsigned short*)(ws + (size_t)B_DIM * D_DIM * 2);    // 32 MB
  float*          beta = (float*)(ws + (size_t)B_DIM * D_DIM * 2
                                     + (size_t)E_DIM * H_DIM * D_DIM * 2);     // 128 KB

  // 1) fp32 -> bf16 conversions
  {
    int n4 = (B_DIM * D_DIM) / 4;
    cvt_bf16_kernel<<<(n4 + 255) / 256, 256, 0, stream>>>(x, xb, n4);
  }
  {
    int n4 = (E_DIM * H_DIM * D_DIM) / 4;
    cvt_bf16_kernel<<<(n4 + 255) / 256, 256, 0, stream>>>(W, wb, n4);
  }

  // 2) fold centers + bias: beta[e,h] = bias[e,h] - c_e . W_eh
  beta_kernel<<<(E_DIM * H_DIM + 255) / 256, 256, 0, stream>>>(centers, W, bias, beta);

  // 3) fused WMMA GEMM + per-expert relu accumulation
  dim3 grid(H_DIM / 128, B_DIM / 128);  // (16, 32)
  moe_relu_sum_kernel<<<grid, 256, 0, stream>>>(xb, wb, beta, out);
}
